// NeuroBranch_13700945674553
// MI455X (gfx1250) — matrix operand order repaired and checked by reading, not compiled, not run on hardware
//
#include <hip/hip_runtime.h>
#include <hip/hip_bf16.h>
#include <math.h>

#define D_        128
#define NVARS     10000
#define NLITS     20000
#define NCLAUSES  50000
#define NEDGES    150000
#define NROUNDS   8
#define EPSN      1e-3f

typedef __attribute__((ext_vector_type(16))) _Float16 v16h;
typedef __attribute__((ext_vector_type(8)))  float    v8f;

// ---------------------------------------------------------------------------
// WMMA helpers (CDNA5: V_WMMA_F32_16X16X32_F16, wave32)
// ---------------------------------------------------------------------------
__device__ __forceinline__ v8f wmma_f16(v16h a, v16h b, v8f c) {
  // 8 args: (neg_a, A, neg_b, B, c_mod, C, reuse_a, reuse_b)
  return __builtin_amdgcn_wmma_f32_16x16x32_f16(false, a, false, b, (short)0, c,
                                                false, false);
}

// A fragment (16x32 f16): lane m (=lane&15) holds row m.
// low half lanes: K = 0..15; high half lanes: K = 16..31.
// rowk points at &X[row][ktile*32]; convert f32 -> f16 on the fly.
__device__ __forceinline__ v16h load_a_f32(const float* __restrict__ rowk, int half16) {
  const float* rp = rowk + (half16 ? 16 : 0);
  v16h a;
#pragma unroll
  for (int i = 0; i < 8; ++i) {
    a[i]     = (_Float16)rp[i];
    a[i + 8] = (_Float16)rp[i + 8];
  }
  return a;
}

__device__ __forceinline__ v16h load_a_lds(const _Float16* __restrict__ rowk, int half16) {
  const _Float16* rp = rowk + (half16 ? 16 : 0);
  v16h a;
#pragma unroll
  for (int i = 0; i < 8; ++i) {
    a[i]     = rp[i];
    a[i + 8] = rp[i + 8];
  }
  return a;
}

// B fragment (32x16 f16): lane holds column n = lane&15; lanes 0-15 cover
// K=0..15 of the tile, lanes 16-31 cover K=16..31. Weights are pre-transposed
// f16 [N=128][K] so each lane reads 16 contiguous halves (32B aligned).
__device__ __forceinline__ v16h load_b_frag(const _Float16* __restrict__ wt,
                                            int ncol, int K, int kcol, int half16) {
  return *(const v16h*)(wt + (size_t)ncol * K + kcol + (half16 ? 16 : 0));
}

// Dense 128->128 layer with A sourced from this wave's LDS tile (f16 [16][128]).
__device__ __forceinline__ void lds_dense_layer(const _Float16* __restrict__ hl,
                                                const _Float16* __restrict__ Wt,
                                                const float* __restrict__ bias,
                                                int m, int half16, v8f acc[8]) {
#pragma unroll
  for (int j = 0; j < 8; ++j) {
    float bv = bias[16 * j + m];
#pragma unroll
    for (int v = 0; v < 8; ++v) acc[j][v] = bv;
  }
#pragma unroll
  for (int kk = 0; kk < 4; ++kk) {
    v16h a = load_a_lds(hl + m * D_ + kk * 32, half16);
#pragma unroll
    for (int j = 0; j < 8; ++j) {
      v16h b = load_b_frag(Wt, 16 * j + m, 128, kk * 32, half16);
      acc[j] = wmma_f16(a, b, acc[j]);
    }
  }
}

// ReLU + transpose C/D layout into this wave's LDS A-tile (f16 [16][128]).
__device__ __forceinline__ void store_relu_lds(_Float16* __restrict__ hl,
                                               const v8f acc[8], int m, int half16) {
#pragma unroll
  for (int j = 0; j < 8; ++j)
#pragma unroll
    for (int v = 0; v < 8; ++v) {
      float x = acc[j][v];
      x = x > 0.f ? x : 0.f;
      hl[(v + half16 * 8) * D_ + 16 * j + m] = (_Float16)x;
    }
}

// Row-wise (x-mean)/(std_unbiased+eps) + residual, stores f32.
__device__ __forceinline__ void normalize_residual_store(const v8f acc[8],
                                                         const float* __restrict__ Xold,
                                                         float* __restrict__ Xout,
                                                         int row0, int m, int half16) {
#pragma unroll
  for (int v = 0; v < 8; ++v) {
    float s = 0.f, q = 0.f;
#pragma unroll
    for (int j = 0; j < 8; ++j) { float x = acc[j][v]; s += x; q += x * x; }
#pragma unroll
    for (int off = 1; off < 16; off <<= 1) {  // 16-lane half-group tree (wave32)
      s += __shfl_xor(s, off, 32);
      q += __shfl_xor(q, off, 32);
    }
    float mean = s * (1.0f / 128.0f);
    float var  = (q - s * mean) * (1.0f / 127.0f);
    var = var > 0.f ? var : 0.f;
    float inv = 1.0f / (sqrtf(var) + EPSN);
    int r = row0 + v + half16 * 8;
#pragma unroll
    for (int j = 0; j < 8; ++j) {
      int col = 16 * j + m;
      Xout[(size_t)r * D_ + col] =
          (acc[j][v] - mean) * inv + Xold[(size_t)r * D_ + col];
    }
  }
}

// ---------------------------------------------------------------------------
// Utility kernels
// ---------------------------------------------------------------------------
__global__ void fill_kernel(float* __restrict__ p, const float* __restrict__ sp, int n) {
  float v = (sp != nullptr) ? *sp : 0.0f;
  for (int i = blockIdx.x * blockDim.x + threadIdx.x; i < n;
       i += gridDim.x * blockDim.x)
    p[i] = v;
}

// src: batch x [K][128] f32 row-major  ->  dst: batch x [128][K] f16
__global__ void convert_wt_kernel(const float* __restrict__ src,
                                  _Float16* __restrict__ dst, int K, int batch) {
  long total = (long)batch * K * D_;
  long stride = (long)gridDim.x * blockDim.x;
  for (long i = (long)blockIdx.x * blockDim.x + threadIdx.x; i < total; i += stride) {
    long b  = i / ((long)K * D_);
    long r  = i - b * (long)K * D_;
    int  k  = (int)(r / D_);
    int  n  = (int)(r - (long)k * D_);
    dst[b * (long)K * D_ + (long)n * K + k] = (_Float16)src[i];
  }
}

// one wave per edge: dst[didx[e]] += scale * src[sidx[e]]  (128 f32 per edge)
__global__ void scatter_add_kernel(const float* __restrict__ src,
                                   float* __restrict__ dst,
                                   const int* __restrict__ sidx,
                                   const int* __restrict__ didx,
                                   const float* __restrict__ scale_p, int n_edges) {
  int gid  = blockIdx.x * blockDim.x + threadIdx.x;
  int e    = gid >> 5;
  int lane = threadIdx.x & 31;
  if (e >= n_edges) return;
  if (e + 16 < n_edges)  // look-ahead prefetch -> global_prefetch_b8
    __builtin_prefetch(src + (size_t)sidx[e + 16] * D_ + lane * 4, 0, 0);
  float scale = *scale_p;
  const float* s = src + (size_t)sidx[e] * D_;
  float*       d = dst + (size_t)didx[e] * D_;
  float4 v = *(const float4*)(s + lane * 4);
  atomicAdd(d + lane * 4 + 0, v.x * scale);
  atomicAdd(d + lane * 4 + 1, v.y * scale);
  atomicAdd(d + lane * 4 + 2, v.z * scale);
  atomicAdd(d + lane * 4 + 3, v.w * scale);
}

// ---------------------------------------------------------------------------
// C update: C_new = normalize(MLP([C | LC])) + C ;  MLP: 256->128->128->128
// ---------------------------------------------------------------------------
__global__ __launch_bounds__(256)
void c_update_kernel(const float* __restrict__ Cin, const float* __restrict__ LC,
                     float* __restrict__ Cout,
                     const _Float16* __restrict__ W0t, const _Float16* __restrict__ W1t,
                     const _Float16* __restrict__ W2t, const float* __restrict__ bias) {
  __shared__ _Float16 hbuf[8][16 * D_];
  const int tid = threadIdx.x;
  const int wave = tid >> 5, lane = tid & 31;
  const int half16 = lane >> 4, m = lane & 15;
  int row0 = (blockIdx.x * 8 + wave) * 16;
  if (row0 > NCLAUSES - 16) row0 = NCLAUSES - 16;
  const int grow = row0 + m;
  _Float16* hl = &hbuf[wave][0];

  // layer 0: K = 256 over [Cin | LC]
  v8f acc[8];
#pragma unroll
  for (int j = 0; j < 8; ++j) {
    float bv = bias[16 * j + m];
#pragma unroll
    for (int v = 0; v < 8; ++v) acc[j][v] = bv;
  }
#pragma unroll
  for (int kk = 0; kk < 8; ++kk) {
    const float* base = (kk < 4) ? Cin : LC;
    int kloc = (kk & 3) * 32;
    v16h a = load_a_f32(base + (size_t)grow * D_ + kloc, half16);
#pragma unroll
    for (int j = 0; j < 8; ++j) {
      v16h b = load_b_frag(W0t, 16 * j + m, 256, kk * 32, half16);
      acc[j] = wmma_f16(a, b, acc[j]);
    }
  }
  store_relu_lds(hl, acc, m, half16);
  __syncthreads();

  lds_dense_layer(hl, W1t, bias + 128, m, half16, acc);   // layer 1
  __syncthreads();
  store_relu_lds(hl, acc, m, half16);
  __syncthreads();

  lds_dense_layer(hl, W2t, bias + 256, m, half16, acc);   // layer 2 (no relu)
  normalize_residual_store(acc, Cin, Cout, row0, m, half16);
}

// ---------------------------------------------------------------------------
// L update: L_new = normalize(MLP([L | CL | L_flip])) + L ; MLP: 384->128->128->128
// ---------------------------------------------------------------------------
__global__ __launch_bounds__(256)
void l_update_kernel(const float* __restrict__ Lin, const float* __restrict__ CL,
                     float* __restrict__ Lout,
                     const _Float16* __restrict__ W0t, const _Float16* __restrict__ W1t,
                     const _Float16* __restrict__ W2t, const float* __restrict__ bias) {
  __shared__ _Float16 hbuf[8][16 * D_];
  const int tid = threadIdx.x;
  const int wave = tid >> 5, lane = tid & 31;
  const int half16 = lane >> 4, m = lane & 15;
  int row0 = (blockIdx.x * 8 + wave) * 16;
  if (row0 > NLITS - 16) row0 = NLITS - 16;
  const int grow = row0 + m;
  const int frow = (grow < NVARS) ? grow + NVARS : grow - NVARS;
  _Float16* hl = &hbuf[wave][0];

  v8f acc[8];
#pragma unroll
  for (int j = 0; j < 8; ++j) {
    float bv = bias[16 * j + m];
#pragma unroll
    for (int v = 0; v < 8; ++v) acc[j][v] = bv;
  }
#pragma unroll
  for (int kk = 0; kk < 12; ++kk) {
    const float* rowk;
    if (kk < 4)      rowk = Lin + (size_t)grow * D_ + kk * 32;
    else if (kk < 8) rowk = CL  + (size_t)grow * D_ + (kk - 4) * 32;
    else             rowk = Lin + (size_t)frow * D_ + (kk - 8) * 32;
    v16h a = load_a_f32(rowk, half16);
#pragma unroll
    for (int j = 0; j < 8; ++j) {
      v16h b = load_b_frag(W0t, 16 * j + m, 384, kk * 32, half16);
      acc[j] = wmma_f16(a, b, acc[j]);
    }
  }
  store_relu_lds(hl, acc, m, half16);
  __syncthreads();

  lds_dense_layer(hl, W1t, bias + 128, m, half16, acc);
  __syncthreads();
  store_relu_lds(hl, acc, m, half16);
  __syncthreads();

  lds_dense_layer(hl, W2t, bias + 256, m, half16, acc);
  normalize_residual_store(acc, Lin, Lout, row0, m, half16);
}

// ---------------------------------------------------------------------------
// V score: V=[L[:nv] | L[nv:]]; relu(V@W0+b0); relu(@W1+b1); @w2+b2 -> [nv]
// ---------------------------------------------------------------------------
__global__ __launch_bounds__(256)
void v_score_kernel(const float* __restrict__ L,
                    const _Float16* __restrict__ W0t, const _Float16* __restrict__ W1t,
                    const float* __restrict__ w2, const float* __restrict__ b0,
                    const float* __restrict__ b1, const float* __restrict__ b2,
                    float* __restrict__ out) {
  __shared__ _Float16 hbuf[8][16 * D_];
  const int tid = threadIdx.x;
  const int wave = tid >> 5, lane = tid & 31;
  const int half16 = lane >> 4, m = lane & 15;
  int row0 = (blockIdx.x * 8 + wave) * 16;
  if (row0 > NVARS - 16) row0 = NVARS - 16;
  const int grow = row0 + m;
  _Float16* hl = &hbuf[wave][0];

  v8f acc[8];
#pragma unroll
  for (int j = 0; j < 8; ++j) {
    float bv = b0[16 * j + m];
#pragma unroll
    for (int v = 0; v < 8; ++v) acc[j][v] = bv;
  }
#pragma unroll
  for (int kk = 0; kk < 8; ++kk) {
    int r = (kk < 4) ? grow : grow + NVARS;
    int kloc = (kk & 3) * 32;
    v16h a = load_a_f32(L + (size_t)r * D_ + kloc, half16);
#pragma unroll
    for (int j = 0; j < 8; ++j) {
      v16h b = load_b_frag(W0t, 16 * j + m, 256, kk * 32, half16);
      acc[j] = wmma_f16(a, b, acc[j]);
    }
  }
  store_relu_lds(hl, acc, m, half16);
  __syncthreads();

  lds_dense_layer(hl, W1t, b1, m, half16, acc);  // layer 1 (pre-relu)

  float b2v = *b2;
  float w2c[8];
#pragma unroll
  for (int j = 0; j < 8; ++j) w2c[j] = w2[16 * j + m];
#pragma unroll
  for (int v = 0; v < 8; ++v) {
    float s = 0.f;
#pragma unroll
    for (int j = 0; j < 8; ++j) {
      float x = acc[j][v];
      x = x > 0.f ? x : 0.f;
      s += x * w2c[j];
    }
#pragma unroll
    for (int off = 1; off < 16; off <<= 1) s += __shfl_xor(s, off, 32);
    if (m == 0) out[row0 + v + half16 * 8] = s + b2v;
  }
}

// ---------------------------------------------------------------------------
// Host orchestration
// ---------------------------------------------------------------------------
extern "C" void kernel_launch(void* const* d_in, const int* in_sizes, int n_in,
                              void* d_out, int out_size, void* d_ws, size_t ws_size,
                              hipStream_t stream) {
  (void)in_sizes; (void)n_in; (void)out_size; (void)ws_size;
  const int*   clause_idx = (const int*)d_in[0];
  const int*   lit_idx    = (const int*)d_in[1];
  // d_in[2], d_in[3] = n_vars, n_clauses (compile-time constants here)
  const float* L_init   = (const float*)d_in[4];
  const float* C_init   = (const float*)d_in[5];
  const float* LC_scale = (const float*)d_in[6];
  const float* CL_scale = (const float*)d_in[7];
  const float* L_W0 = (const float*)d_in[8];
  const float* L_Wr = (const float*)d_in[9];
  const float* L_b  = (const float*)d_in[10];
  const float* C_W0 = (const float*)d_in[11];
  const float* C_Wr = (const float*)d_in[12];
  const float* C_b  = (const float*)d_in[13];
  const float* V_W0 = (const float*)d_in[14];
  const float* V_W1 = (const float*)d_in[15];
  const float* V_W2 = (const float*)d_in[16];
  const float* V_b0 = (const float*)d_in[17];
  const float* V_b1 = (const float*)d_in[18];
  const float* V_b2 = (const float*)d_in[19];
  float* out = (float*)d_out;

  char* w = (char*)d_ws;
  auto take = [&](size_t bytes) -> char* {
    char* p = w;
    w += (bytes + 255) & ~(size_t)255;
    return p;
  };
  float* Lb0 = (float*)take((size_t)NLITS * D_ * 4);
  float* Lb1 = (float*)take((size_t)NLITS * D_ * 4);
  float* Cb0 = (float*)take((size_t)NCLAUSES * D_ * 4);
  float* Cb1 = (float*)take((size_t)NCLAUSES * D_ * 4);
  float* LC  = (float*)take((size_t)NCLAUSES * D_ * 4);
  float* CL  = (float*)take((size_t)NLITS * D_ * 4);
  _Float16* cw0 = (_Float16*)take((size_t)8 * 256 * D_ * 2);
  _Float16* cwr = (_Float16*)take((size_t)16 * D_ * D_ * 2);
  _Float16* lw0 = (_Float16*)take((size_t)8 * 384 * D_ * 2);
  _Float16* lwr = (_Float16*)take((size_t)16 * D_ * D_ * 2);
  _Float16* vw0 = (_Float16*)take((size_t)256 * D_ * 2);
  _Float16* vw1 = (_Float16*)take((size_t)D_ * D_ * 2);

  // One-time (per call) weight convert+transpose to f16 [N][K]
  convert_wt_kernel<<<512, 256, 0, stream>>>(C_W0, cw0, 256, 8);
  convert_wt_kernel<<<512, 256, 0, stream>>>(C_Wr, cwr, 128, 16);
  convert_wt_kernel<<<512, 256, 0, stream>>>(L_W0, lw0, 384, 8);
  convert_wt_kernel<<<512, 256, 0, stream>>>(L_Wr, lwr, 128, 16);
  convert_wt_kernel<<<128, 256, 0, stream>>>(V_W0, vw0, 256, 1);
  convert_wt_kernel<<<128, 256, 0, stream>>>(V_W1, vw1, 128, 1);

  fill_kernel<<<1024, 256, 0, stream>>>(Lb0, L_init, NLITS * D_);
  fill_kernel<<<1024, 256, 0, stream>>>(Cb0, C_init, NCLAUSES * D_);

  float *Lcur = Lb0, *Lnext = Lb1, *Ccur = Cb0, *Cnext = Cb1;
  const int scatter_blocks = (NEDGES * 32 + 255) / 256;
  for (int t = 0; t < NROUNDS; ++t) {
    fill_kernel<<<1024, 256, 0, stream>>>(LC, nullptr, NCLAUSES * D_);
    scatter_add_kernel<<<scatter_blocks, 256, 0, stream>>>(
        Lcur, LC, lit_idx, clause_idx, LC_scale, NEDGES);
    c_update_kernel<<<(NCLAUSES + 127) / 128, 256, 0, stream>>>(
        Ccur, LC, Cnext,
        cw0 + (size_t)t * 256 * D_,
        cwr + (size_t)(2 * t) * D_ * D_,
        cwr + (size_t)(2 * t + 1) * D_ * D_,
        C_b + (size_t)t * 3 * D_);
    fill_kernel<<<1024, 256, 0, stream>>>(CL, nullptr, NLITS * D_);
    scatter_add_kernel<<<scatter_blocks, 256, 0, stream>>>(
        Cnext, CL, clause_idx, lit_idx, CL_scale, NEDGES);
    l_update_kernel<<<(NLITS + 127) / 128, 256, 0, stream>>>(
        Lcur, CL, Lnext,
        lw0 + (size_t)t * 384 * D_,
        lwr + (size_t)(2 * t) * D_ * D_,
        lwr + (size_t)(2 * t + 1) * D_ * D_,
        L_b + (size_t)t * 3 * D_);
    float* tl = Lcur; Lcur = Lnext; Lnext = tl;
    float* tc = Ccur; Ccur = Cnext; Cnext = tc;
  }
  v_score_kernel<<<(NVARS + 127) / 128, 256, 0, stream>>>(
      Lcur, vw0, vw1, V_W2, V_b0, V_b1, V_b2, out);
}
